// SingularformerGLUDecoderBlock_86844238725668
// MI455X (gfx1250) — matrix operand order repaired and checked by reading, not compiled, not run on hardware
//
#include <hip/hip_runtime.h>
#include <hip/hip_bf16.h>

// ---------------------------------------------------------------------------
// Singularformer GLU decoder block for MI455X (gfx1250, wave32, WMMA).
// Heavy matmuls via v_wmma_f32_16x16x32_bf16 (f32 accumulate); LDS tiles are
// stored in fragment order so fragment IO is ds_load_b128/ds_store_b128.
// ---------------------------------------------------------------------------

typedef __bf16 bf16_t;
typedef __attribute__((ext_vector_type(16))) __bf16 v16bf;
typedef __attribute__((ext_vector_type(8)))  __bf16 v8bf;
typedef __attribute__((ext_vector_type(8)))  float  v8f;

#define BDIM   8
#define LSEQ   2048
#define CDIM   512
#define HDIM   1024
#define RDIM   64
#define DKDIM  64
#define NCH    32           // cumsum chunks
#define CHL    (LSEQ/NCH)   // 64 rows per chunk

__device__ __forceinline__ float fast_sigmoid(float x)
{
    return __builtin_amdgcn_rcpf(1.f + __expf(-x));
}

// ---------------------------------------------------------------------------
// Generic bf16-WMMA GEMM:  C[b] = act( scale * A[b] @ B[b] + bias )
//   A(m,k) at A + b*sAb + m*sAr + k*sAc   (f32 -> bf16)
//   B(k,n) at B + b*sBb + k*sBr + n*sBc
//   C row-major [M,N] per batch (stride sCb between batches)
// Template: MT = M-tile multiplier (tile is 64*MT x 64), ACONTIG = (sAc==1).
// Requires M%(64*MT)==0, N%64==0, K%32==0 (true for every call here).
// Block: 256 threads (8 wave32). K-step 32.
// Wave w -> M subtiles (w>>1) + 4*c (c<MT), N subtiles 2*(w&1), 2*(w&1)+1:
//   2*MT WMMAs per wave per K-step.
//
// LDS tiles live in *fragment order*: [sub16][eGrp][lane][8 bf16], so every
// lane's fragment half is one 16B ds_load_b128 and every staging thread
// writes exactly one 16B ds_store_b128 per slot.
// Fragment mapping (ISA 7.12.2, 16-bit A 16x32 / B 32x16):
//   A: lane (h=lane>>4, r=lane&15), elem e<8 -> K=8h+e ; e>=8 -> K=16+8h+(e-8)
//   B: lane -> column r, elem e -> K as for A (e<8 -> 8h+e ; e>=8 -> 16+8h+(e-8))
// ---------------------------------------------------------------------------
template<int MT, bool ACONTIG>
__global__ void sgf_gemm(const float* __restrict__ A, long sAr, long sAc, long sAb,
                         const float* __restrict__ B, long sBr, long sBc, long sBb,
                         float* __restrict__ C, long sCb,
                         int N, int K,
                         const float* __restrict__ bias, float scale, int act)
{
    __shared__ __align__(16) bf16_t Afrag[4 * MT][2][32][8]; // [Msub][eGrp][lane][8]
    __shared__ __align__(16) bf16_t Bfrag[4][2][32][8];      // [Nsub][eGrp][lane][8]

    const int tid  = threadIdx.x;
    const int lane = tid & 31;
    const int wave = tid >> 5;
    const int wmb  = wave >> 1;          // 0..3 : base M subtile
    const int wn   = wave & 1;           // 0..1 : N pair (subtiles 2wn, 2wn+1)
    const int half = lane >> 4;
    const int r    = lane & 15;

    const long tileM = (long)blockIdx.y * (64 * MT);
    const long tileN = (long)blockIdx.x * 64;
    const float* Ab = A + (long)blockIdx.z * sAb;
    const float* Bb = B + (long)blockIdx.z * sBb;
    float*       Cb = C + (long)blockIdx.z * sCb;

    // ---- A staging slots (MT per thread): idx = row*4 + chunk
    const float* aptr[MT];
    int a_s[MT], a_g[MT], a_l[MT];
#pragma unroll
    for (int c = 0; c < MT; ++c) {
        int idx = c * 256 + tid;
        int row = idx >> 2;              // 0..64*MT-1
        int h   = idx & 1;
        int g   = (idx >> 1) & 1;
        a_s[c]  = row >> 4;
        a_g[c]  = g;
        a_l[c]  = (h << 4) | (row & 15);
        int kb  = 8 * h + 16 * g;        // K base of this 8-run
        aptr[c] = Ab + (tileM + row) * sAr + (ACONTIG ? (long)kb : (long)kb * sAc);
    }
    const long aStep = ACONTIG ? 32 : 32 * sAc;

    // ---- B staging slot: tid = (half*2+g)*64 + (col 0..63); coalesced cols
    const int b_G    = tid >> 6;
    const int b_half = b_G >> 1;
    const int b_g    = b_G & 1;
    const int b_col  = tid & 63;
    const int b_sn   = b_col >> 4;
    const int b_l    = (b_half << 4) | (b_col & 15);
    const int b_kb   = 8 * b_half + 16 * b_g;
    const float* bptr = Bb + (tileN + b_col) * sBc + (long)b_kb * sBr;
    const long bStep  = 32 * sBr;

    const v8f vzero = {};
    v8f acc[MT][2];
#pragma unroll
    for (int c = 0; c < MT; ++c) { acc[c][0] = vzero; acc[c][1] = vzero; }

    for (int k0 = 0; k0 < K; k0 += 32) {
        // Stage A: 8 K-consecutive f32 -> one 16B LDS store per slot
#pragma unroll
        for (int c = 0; c < MT; ++c) {
            v8bf apk;
            if (ACONTIG) {
                const float4* p = (const float4*)aptr[c];
                float4 f0 = p[0], f1 = p[1];
                apk[0] = (bf16_t)f0.x; apk[1] = (bf16_t)f0.y;
                apk[2] = (bf16_t)f0.z; apk[3] = (bf16_t)f0.w;
                apk[4] = (bf16_t)f1.x; apk[5] = (bf16_t)f1.y;
                apk[6] = (bf16_t)f1.z; apk[7] = (bf16_t)f1.w;
            } else {
#pragma unroll
                for (int j = 0; j < 8; ++j)
                    apk[j] = (bf16_t)aptr[c][(long)j * sAc];
            }
            *(v8bf*)&Afrag[a_s[c]][a_g[c]][a_l[c]][0] = apk;
            aptr[c] += aStep;
        }

        // Stage B: 8 K-strided f32 (coalesced across 64-thread column sweep)
        v8bf bpk;
#pragma unroll
        for (int j = 0; j < 8; ++j)
            bpk[j] = (bf16_t)bptr[(long)j * sBr];
        *(v8bf*)&Bfrag[b_sn][b_g][b_l][0] = bpk;
        bptr += bStep;

        __syncthreads();

        // Fragment loads: 16B vector LDS reads
        v8bf b0lo = *(const v8bf*)&Bfrag[2 * wn][0][lane][0];
        v8bf b0hi = *(const v8bf*)&Bfrag[2 * wn][1][lane][0];
        v8bf b1lo = *(const v8bf*)&Bfrag[2 * wn + 1][0][lane][0];
        v8bf b1hi = *(const v8bf*)&Bfrag[2 * wn + 1][1][lane][0];
        v16bf bf0 = __builtin_shufflevector(b0lo, b0hi,
                        0,1,2,3,4,5,6,7,8,9,10,11,12,13,14,15);
        v16bf bf1 = __builtin_shufflevector(b1lo, b1hi,
                        0,1,2,3,4,5,6,7,8,9,10,11,12,13,14,15);

#pragma unroll
        for (int c = 0; c < MT; ++c) {
            v8bf alo = *(const v8bf*)&Afrag[wmb + 4 * c][0][lane][0];
            v8bf ahi = *(const v8bf*)&Afrag[wmb + 4 * c][1][lane][0];
            v16bf af = __builtin_shufflevector(alo, ahi,
                           0,1,2,3,4,5,6,7,8,9,10,11,12,13,14,15);
            acc[c][0] = __builtin_amdgcn_wmma_f32_16x16x32_bf16(
                            false, af, false, bf0, (short)0, acc[c][0], false, false);
            acc[c][1] = __builtin_amdgcn_wmma_f32_16x16x32_bf16(
                            false, af, false, bf1, (short)0, acc[c][1], false, false);
        }
        __syncthreads();
    }

    // C/D layout: VGPR e, lanes 0-15 -> M=e,N=lane; lanes 16-31 -> M=8+e,N=lane-16
#pragma unroll
    for (int c = 0; c < MT; ++c) {
#pragma unroll
        for (int e = 0; e < 8; ++e) {
            long row = tileM + (wmb + 4 * c) * 16 + half * 8 + e;
            long c0  = tileN + wn * 32 + r;
            long c1  = c0 + 16;
            float y0 = acc[c][0][e] * scale;
            float y1 = acc[c][1][e] * scale;
            if (bias) { y0 += bias[c0]; y1 += bias[c1]; }
            if (act == 1) {   // SiLU
                y0 = y0 * fast_sigmoid(y0);
                y1 = y1 * fast_sigmoid(y1);
            }
            Cb[row * N + c0] = y0;
            Cb[row * N + c1] = y1;
        }
    }
}

// ---------------------------------------------------------------------------
// Cumulative running mean vc = cumsum(qx, axis=L) / (l+1), 3-phase chunked scan
// ---------------------------------------------------------------------------
__global__ void sgf_part(const float* __restrict__ qx, float* __restrict__ part)
{
    int c = blockIdx.x * 256 + threadIdx.x;
    int j = blockIdx.y;
    int b = blockIdx.z;
    size_t base = ((size_t)b * LSEQ + (size_t)j * CHL) * CDIM + c;
    float s = 0.f;
    for (int l = 0; l < CHL; ++l) s += qx[base + (size_t)l * CDIM];
    part[((size_t)b * NCH + j) * CDIM + c] = s;
}

__global__ void sgf_scan(float* __restrict__ part)
{
    int c = blockIdx.x * 256 + threadIdx.x;
    int b = blockIdx.y;
    float run = 0.f;
    for (int j = 0; j < NCH; ++j) {
        size_t idx = ((size_t)b * NCH + j) * CDIM + c;
        float t = part[idx];
        part[idx] = run;          // exclusive prefix
        run += t;
    }
}

__global__ void sgf_apply(const float* __restrict__ qx, const float* __restrict__ part,
                          float* __restrict__ vc)
{
    int c = blockIdx.x * 256 + threadIdx.x;
    int j = blockIdx.y;
    int b = blockIdx.z;
    float acc = part[((size_t)b * NCH + j) * CDIM + c];
    size_t base = ((size_t)b * LSEQ + (size_t)j * CHL) * CDIM + c;
    for (int l = 0; l < CHL; ++l) {
        acc += qx[base + (size_t)l * CDIM];
        vc[base + (size_t)l * CDIM] = acc / (float)(j * CHL + l + 1);
    }
}

// ---------------------------------------------------------------------------
// z[b,l,:] = sigmoid( q[b,l]·k[b,l-1] per head / sqrt(DK) ) * v[b,l-1,:], row0=0
// One workgroup per (b,l); wave m handles head m (DK=64 over 32 lanes).
// ---------------------------------------------------------------------------
__global__ void sgf_scores_z(const float* __restrict__ q, const float* __restrict__ k,
                             const float* __restrict__ v, float* __restrict__ z)
{
    int l = blockIdx.x, b = blockIdx.y, t = threadIdx.x;
    size_t rowq = ((size_t)b * LSEQ + l) * CDIM;
    if (l == 0) { z[rowq + t] = 0.f; z[rowq + t + 256] = 0.f; return; }
    size_t rowk = rowq - CDIM;
    int wave = t >> 5, lane = t & 31;
    int base = wave * DKDIM;
    float p = q[rowq + base + lane]      * k[rowk + base + lane]
            + q[rowq + base + lane + 32] * k[rowk + base + lane + 32];
#pragma unroll
    for (int off = 16; off >= 1; off >>= 1) p += __shfl_xor(p, off, 32);
    float s = fast_sigmoid(p * 0.125f);   // /sqrt(64)
    z[rowq + base + lane]      = s * v[rowk + base + lane];
    z[rowq + base + lane + 32] = s * v[rowk + base + lane + 32];
}

// ---------------------------------------------------------------------------
// out = LayerNorm(x + z) with affine (C=512, 256 threads x 2 elems)
// ---------------------------------------------------------------------------
__global__ void sgf_add_ln(const float* __restrict__ x, const float* __restrict__ z,
                           const float* __restrict__ g, const float* __restrict__ beta,
                           float* __restrict__ out)
{
    int t = threadIdx.x;
    size_t base = (size_t)blockIdx.x * CDIM;
    float x0 = x[base + t]       + z[base + t];
    float x1 = x[base + t + 256] + z[base + t + 256];
    __shared__ float s_sum[256], s_sq[256];
    s_sum[t] = x0 + x1;
    s_sq[t]  = x0 * x0 + x1 * x1;
    __syncthreads();
    for (int o = 128; o > 0; o >>= 1) {
        if (t < o) { s_sum[t] += s_sum[t + o]; s_sq[t] += s_sq[t + o]; }
        __syncthreads();
    }
    float mean = s_sum[0] * (1.f / CDIM);
    float var  = s_sq[0]  * (1.f / CDIM) - mean * mean;
    float rs   = rsqrtf(var + 1e-5f);
    out[base + t]       = (x0 - mean) * rs * g[t]       + beta[t];
    out[base + t + 256] = (x1 - mean) * rs * g[t + 256] + beta[t + 256];
}

// ---------------------------------------------------------------------------
// Masked softmax over L (axis=1) of pScore [B,L,R], in place. Block per (b,r).
// ---------------------------------------------------------------------------
__global__ void sgf_softmax_L(float* __restrict__ pS, const int* __restrict__ mask)
{
    int r = blockIdx.x, b = blockIdx.y, t = threadIdx.x;
    const int* mp = mask + (size_t)b * LSEQ;
    float* col = pS + (size_t)b * LSEQ * RDIM + r;
    __shared__ float sh[256];

    float mx = -3.4e38f;
    for (int l = t; l < LSEQ; l += 256) {
        float v = (mp[l] == 0) ? -32767.f : col[(size_t)l * RDIM];
        mx = fmaxf(mx, v);
    }
    sh[t] = mx; __syncthreads();
    for (int o = 128; o > 0; o >>= 1) { if (t < o) sh[t] = fmaxf(sh[t], sh[t + o]); __syncthreads(); }
    mx = sh[0]; __syncthreads();

    float sum = 0.f;
    for (int l = t; l < LSEQ; l += 256) {
        float v = (mp[l] == 0) ? -32767.f : col[(size_t)l * RDIM];
        sum += __expf(v - mx);
    }
    sh[t] = sum; __syncthreads();
    for (int o = 128; o > 0; o >>= 1) { if (t < o) sh[t] += sh[t + o]; __syncthreads(); }
    float inv = 1.f / sh[0];

    for (int l = t; l < LSEQ; l += 256) {
        float v = (mp[l] == 0) ? -32767.f : col[(size_t)l * RDIM];
        col[(size_t)l * RDIM] = __expf(v - mx) * inv;
    }
}

// Softmax over last dim R=64, in place. Block of 64 threads per row.
__global__ void sgf_softmax_R(float* __restrict__ S)
{
    int t = threadIdx.x;
    float* p = S + (size_t)blockIdx.x * RDIM;
    float v = p[t];
    __shared__ float sh[64];
    sh[t] = v; __syncthreads();
    for (int o = 32; o > 0; o >>= 1) { if (t < o) sh[t] = fmaxf(sh[t], sh[t + o]); __syncthreads(); }
    float mx = sh[0]; __syncthreads();
    float e = __expf(v - mx);
    sh[t] = e; __syncthreads();
    for (int o = 32; o > 0; o >>= 1) { if (t < o) sh[t] += sh[t + o]; __syncthreads(); }
    p[t] = e / sh[0];
}

__global__ void sgf_mul(float* __restrict__ z2, const float* __restrict__ g, long n)
{
    long i = (long)blockIdx.x * blockDim.x + threadIdx.x;
    long stride = (long)gridDim.x * blockDim.x;
    for (; i < n; i += stride) z2[i] *= g[i];
}

// ---------------------------------------------------------------------------
// Host side
// ---------------------------------------------------------------------------
static inline void launch_gemm(hipStream_t s,
                               const float* A, long sAr, long sAc, long sAb,
                               const float* B, long sBr, long sBc, long sBb,
                               float* C, long sCb,
                               int M, int N, int K, int batch,
                               const float* bias, float scale, int act)
{
    dim3 blk(256);
    const bool ac = (sAc == 1);
    if (M % 128 == 0) {
        dim3 grid(N / 64, M / 128, batch);
        if (ac)
            sgf_gemm<2, true ><<<grid, blk, 0, s>>>(A, sAr, sAc, sAb, B, sBr, sBc, sBb,
                                                    C, sCb, N, K, bias, scale, act);
        else
            sgf_gemm<2, false><<<grid, blk, 0, s>>>(A, sAr, sAc, sAb, B, sBr, sBc, sBb,
                                                    C, sCb, N, K, bias, scale, act);
    } else {
        dim3 grid(N / 64, M / 64, batch);
        if (ac)
            sgf_gemm<1, true ><<<grid, blk, 0, s>>>(A, sAr, sAc, sAb, B, sBr, sBc, sBb,
                                                    C, sCb, N, K, bias, scale, act);
        else
            sgf_gemm<1, false><<<grid, blk, 0, s>>>(A, sAr, sAc, sAb, B, sBr, sBc, sBb,
                                                    C, sCb, N, K, bias, scale, act);
    }
}

extern "C" void kernel_launch(void* const* d_in, const int* in_sizes, int n_in,
                              void* d_out, int out_size, void* d_ws, size_t ws_size,
                              hipStream_t stream)
{
    const float* qx     = (const float*)d_in[0];
    const float* vx     = (const float*)d_in[1];
    const int*   maskP  = (const int*)  d_in[2];
    const float* WQ1    = (const float*)d_in[3];
    const float* WK1    = (const float*)d_in[4];
    const float* WV1    = (const float*)d_in[5];
    const float* WO1    = (const float*)d_in[6];
    const float* qln_g  = (const float*)d_in[7];
    const float* qln_b  = (const float*)d_in[8];
    const float* Wp     = (const float*)d_in[9];
    const float* bp     = (const float*)d_in[10];
    const float* Wq2    = (const float*)d_in[11];
    const float* bq2    = (const float*)d_in[12];
    const float* Wk2    = (const float*)d_in[13];
    const float* bk2    = (const float*)d_in[14];
    const float* Wv2    = (const float*)d_in[15];
    const float* bv2    = (const float*)d_in[16];
    const float* Wu     = (const float*)d_in[17];
    const float* bu     = (const float*)d_in[18];
    const float* Wo2    = (const float*)d_in[19];
    const float* bo2    = (const float*)d_in[20];
    const float* ln_g   = (const float*)d_in[21];
    const float* ln_b   = (const float*)d_in[22];
    float* out = (float*)d_out;

    const size_t BL = (size_t)BDIM * LSEQ;   // 16384

    // Workspace carve-up (floats); buffers reused across phases.
    float* w = (float*)d_ws;
    float* bufA    = w; w += BL * CDIM;            // vc, then zin
    float* bufB    = w; w += BL * CDIM;            // q,  then zout
    float* bufC    = w; w += BL * CDIM;            // k,  then qx2
    float* bufD    = w; w += BL * CDIM;            // v,  then y
    float* bufGate = w; w += BL * HDIM;            // silu(qx2@Wu+bu)
    float* bufZ2   = w; w += BL * HDIM;            // z2 (then gate*z2)
    float* bufPS   = w; w += BL * RDIM;            // pScore/pAlpha
    float* bufPKV  = w; w += (size_t)BDIM * RDIM * CDIM;
    float* bufQ2   = w; w += BL * DKDIM;           // queries
    float* bufKeys = w; w += (size_t)BDIM * RDIM * DKDIM;
    float* bufVals = w; w += (size_t)BDIM * RDIM * HDIM;
    float* bufS2   = w; w += BL * RDIM;            // scores2/alpha
    float* bufPart = w; w += (size_t)BDIM * NCH * CDIM;
    (void)ws_size; (void)in_sizes; (void)n_in; (void)out_size;

    // 1) vc = cumsum(qx)/div  (chunked scan)
    sgf_part <<<dim3(CDIM/256, NCH, BDIM), 256, 0, stream>>>(qx, bufPart);
    sgf_scan <<<dim3(CDIM/256, BDIM),      256, 0, stream>>>(bufPart);
    sgf_apply<<<dim3(CDIM/256, NCH, BDIM), 256, 0, stream>>>(qx, bufPart, bufA);

    // 2) q = qx@WQ1 ; k = vc@WK1 ; v = vc@WV1      [BL,512]
    launch_gemm(stream, qx,   CDIM,1,0, WQ1, CDIM,1,0, bufB, 0, (int)BL, CDIM, CDIM, 1, nullptr, 1.f, 0);
    launch_gemm(stream, bufA, CDIM,1,0, WK1, CDIM,1,0, bufC, 0, (int)BL, CDIM, CDIM, 1, nullptr, 1.f, 0);
    launch_gemm(stream, bufA, CDIM,1,0, WV1, CDIM,1,0, bufD, 0, (int)BL, CDIM, CDIM, 1, nullptr, 1.f, 0);

    // 3) zin = sigmoid(q·k_prev / 8) * v_prev (shifted), row 0 = 0   -> bufA
    sgf_scores_z<<<dim3(LSEQ, BDIM), 256, 0, stream>>>(bufB, bufC, bufD, bufA);

    // 4) zout = zin @ WO1  -> bufB
    launch_gemm(stream, bufA, CDIM,1,0, WO1, CDIM,1,0, bufB, 0, (int)BL, CDIM, CDIM, 1, nullptr, 1.f, 0);

    // 5) qx2 = LN(qx + zout) -> bufC
    sgf_add_ln<<<(int)BL, 256, 0, stream>>>(qx, bufB, qln_g, qln_b, bufC);

    // 6) gate = silu(qx2@Wu + bu)  [BL,1024]
    launch_gemm(stream, bufC, CDIM,1,0, Wu, HDIM,1,0, bufGate, 0, (int)BL, HDIM, CDIM, 1, bu, 1.f, 1);

    // 7) pScore = vx@Wp + bp  [BL,64]; masked softmax over L (in place)
    launch_gemm(stream, vx, CDIM,1,0, Wp, RDIM,1,0, bufPS, 0, (int)BL, RDIM, CDIM, 1, bp, 1.f, 0);
    sgf_softmax_L<<<dim3(RDIM, BDIM), 256, 0, stream>>>(bufPS, maskP);

    // 8) pkvx[b] = pAlpha[b]^T @ vx[b]   [B,64,512]  (A transposed via strides)
    launch_gemm(stream, bufPS, 1, RDIM, (long)LSEQ*RDIM,
                        vx,    CDIM, 1, (long)LSEQ*CDIM,
                        bufPKV, (long)RDIM*CDIM,
                        RDIM, CDIM, LSEQ, BDIM, nullptr, 1.f, 0);

    // 9) queries/keys/values (SiLU + bias fused)
    launch_gemm(stream, bufC,   CDIM,1,0, Wq2, DKDIM,1,0, bufQ2,   0, (int)BL,   DKDIM, CDIM, 1, bq2, 1.f, 1);
    launch_gemm(stream, bufPKV, CDIM,1,0, Wk2, DKDIM,1,0, bufKeys, 0, BDIM*RDIM, DKDIM, CDIM, 1, bk2, 1.f, 1);
    launch_gemm(stream, bufPKV, CDIM,1,0, Wv2, HDIM,1,0,  bufVals, 0, BDIM*RDIM, HDIM,  CDIM, 1, bv2, 1.f, 1);

    // 10) scores2[b] = queries[b] @ keys[b]^T / 8 ; softmax over R (in place)
    launch_gemm(stream, bufQ2,   DKDIM, 1, (long)LSEQ*DKDIM,
                        bufKeys, 1, DKDIM, (long)RDIM*DKDIM,
                        bufS2,   (long)LSEQ*RDIM,
                        LSEQ, RDIM, DKDIM, BDIM, nullptr, 0.125f, 0);
    sgf_softmax_R<<<(int)BL, 64, 0, stream>>>(bufS2);

    // 11) z2[b] = alpha[b] @ values[b]  [B,L,1024]; then z2 *= gate
    launch_gemm(stream, bufS2,   RDIM, 1, (long)LSEQ*RDIM,
                        bufVals, HDIM, 1, (long)RDIM*HDIM,
                        bufZ2,   (long)LSEQ*HDIM,
                        LSEQ, HDIM, RDIM, BDIM, nullptr, 1.f, 0);
    sgf_mul<<<4096, 256, 0, stream>>>(bufZ2, bufGate, (long)BL * HDIM);

    // 12) y = (gate*z2) @ Wo2 + bo2  -> bufD ; out = LN(qx2 + y)
    launch_gemm(stream, bufZ2, HDIM,1,0, Wo2, CDIM,1,0, bufD, 0, (int)BL, CDIM, HDIM, 1, bo2, 1.f, 0);
    sgf_add_ln<<<(int)BL, 256, 0, stream>>>(bufC, bufD, ln_g, ln_b, out);
}
